// AliBiMultiHeadAttention_42322607735295
// MI455X (gfx1250) — matrix operand order repaired and checked by reading, not compiled, not run on hardware
//
#include <hip/hip_runtime.h>

// ---------------------------------------------------------------------------
// ALiBi multi-head attention forward for MI455X (gfx1250, wave32, WMMA).
// All matmuls use v_wmma_f32_16x16x32_bf16 (fp32 accumulate); softmax in fp32.
// d_out = [ out (B*T*C f32) | attn (B*H*T*T f32) ].
// ---------------------------------------------------------------------------

typedef __bf16 bf16;
typedef __attribute__((ext_vector_type(16))) __bf16 v16bf;
typedef __attribute__((ext_vector_type(8)))  __bf16 v8bf;
typedef __attribute__((ext_vector_type(4)))  __bf16 v4bf;
typedef __attribute__((ext_vector_type(8)))  float  v8f;

#define B_  2
#define T_  2048
#define C_  2048
#define H_  16
#define HD_ 128
#define M_  (B_ * T_)                     // 4096 rows of x
#define SCALE 0.08838834764831845f       // 1/sqrt(128)

static __device__ __forceinline__ v8f wmma_bf16(v16bf a, v16bf b, v8f c) {
  return __builtin_amdgcn_wmma_f32_16x16x32_bf16(
      /*neg_a=*/false, a, /*neg_b=*/false, b,
      /*c_mod=*/(short)0, c, /*reuse_a=*/false, /*reuse_b=*/false);
}

// A-fragment (16x32 bf16), row-major source, tile points at (row0,k0).
// ISA layout: lane<16 holds row m=lane, K in {0..7,16..23}; lane>=16: K in {8..15,24..31}.
static __device__ __forceinline__ v16bf load_a_bf16(const bf16* tile, int ld) {
  int lane = threadIdx.x & 31;
  int m  = lane & 15;
  int kb = (lane >> 4) << 3;                       // 0 or 8
  const bf16* p = tile + (size_t)m * ld + kb;
  v8bf lo = *(const v8bf*)p;                       // K = kb .. kb+7
  v8bf hi = *(const v8bf*)(p + 16);                // K = 16+kb .. 16+kb+7
  return __builtin_shufflevector(lo, hi, 0,1,2,3,4,5,6,7,8,9,10,11,12,13,14,15);
}

// Same A-fragment but from an fp32 source (attn), converting to bf16.
static __device__ __forceinline__ v16bf load_a_f32(const float* tile, int ld) {
  int lane = threadIdx.x & 31;
  int m  = lane & 15;
  int kb = (lane >> 4) << 3;
  const float* p = tile + (size_t)m * ld + kb;
  v16bf r;
#pragma unroll
  for (int i = 0; i < 8; ++i) r[i] = (bf16)p[i];
#pragma unroll
  for (int i = 0; i < 8; ++i) r[8 + i] = (bf16)p[16 + i];
  return r;
}

// B-fragment (32x16 bf16). Source is "column-per-row" storage: column n of B
// lives at tile + n*ld, contiguous over K (e.g. W row-major N x K).
// ISA layout: lane holds column n=lane&15; lanes<16 K in {0..7,16..23}, lanes>=16 K in {8..15,24..31}.
static __device__ __forceinline__ v16bf load_b_bf16(const bf16* tile, int ld) {
  int lane = threadIdx.x & 31;
  int n  = lane & 15;
  int kb = (lane >> 4) << 3;                       // 0 or 8
  const bf16* p = tile + (size_t)n * ld + kb;
  v8bf lo = *(const v8bf*)p;
  v8bf hi = *(const v8bf*)(p + 16);
  return __builtin_shufflevector(lo, hi, 0,1,2,3,4,5,6,7,8,9,10,11,12,13,14,15);
}

// D-fragment store, row-major bf16. VGPR v holds row (v + 8*(lane>>4)), col lane&15.
// Each store_b16 is two 32B-contiguous lane groups -> coalesces well.
static __device__ __forceinline__ void store_d_bf16(bf16* out, int ld, const v8f& c) {
  int lane = threadIdx.x & 31;
  int n  = lane & 15;
  int mo = (lane >> 4) << 3;
#pragma unroll
  for (int v = 0; v < 8; ++v)
    out[(size_t)(mo + v) * ld + n] = (bf16)c[v];
}

// ---------------------------------------------------------------------------
// K0: fp32 -> bf16 conversion (x and the four weight matrices), 4 elems/thread
// ---------------------------------------------------------------------------
__global__ __launch_bounds__(256) void cvt_f32_bf16(const float* __restrict__ in,
                                                    bf16* __restrict__ out, int n) {
  int i = (blockIdx.x * 256 + threadIdx.x) * 4;
  if (i + 3 < n) {
    float4 f = *(const float4*)(in + i);
    v4bf pk;
    pk[0] = (bf16)f.x; pk[1] = (bf16)f.y; pk[2] = (bf16)f.z; pk[3] = (bf16)f.w;
    *(v4bf*)(out + i) = pk;
  }
}

// ---------------------------------------------------------------------------
// K1: y = x @ W^T  (M_ x C_ x C_), bf16 in/out, f32 accum.
// Block = 256 thr (8 waves), block tile 128x256, wave tile 64x64 (4x4 frags).
// vmode==1 stores the result head-transposed as vT: (B,H,HD,T) -- the
// D-fragment's 8 rows per lane are contiguous in t -> one 16B store.
// ---------------------------------------------------------------------------
__global__ __launch_bounds__(256) void gemm_xwt_bf16(const bf16* __restrict__ A,
                                                     const bf16* __restrict__ W,
                                                     bf16* __restrict__ out, int vmode) {
  const int K = C_, N = C_;
  int wid = threadIdx.x >> 5;
  int wm = wid & 1, wn = wid >> 1;                 // 2 waves M x 4 waves N
  int row0 = blockIdx.x * 128 + wm * 64;
  int col0 = blockIdx.y * 256 + wn * 64;
  v8f acc[4][4] = {};
#pragma unroll 1
  for (int k0 = 0; k0 < K; k0 += 32) {
    v16bf a[4];
#pragma unroll
    for (int mt = 0; mt < 4; ++mt)
      a[mt] = load_a_bf16(A + (size_t)(row0 + mt * 16) * K + k0, K);
#pragma unroll
    for (int nt = 0; nt < 4; ++nt) {
      v16bf b = load_b_bf16(W + (size_t)(col0 + nt * 16) * K + k0, K);
#pragma unroll
      for (int mt = 0; mt < 4; ++mt)
        acc[mt][nt] = wmma_bf16(a[mt], b, acc[mt][nt]);
    }
  }
  int lane = threadIdx.x & 31;
  if (vmode == 0) {
#pragma unroll
    for (int mt = 0; mt < 4; ++mt)
#pragma unroll
      for (int nt = 0; nt < 4; ++nt)
        store_d_bf16(out + (size_t)(row0 + mt * 16) * N + col0 + nt * 16, N,
                     acc[mt][nt]);
  } else {
    // v output -> vT layout (B,H,HD,T)
#pragma unroll
    for (int mt = 0; mt < 4; ++mt) {
      int tbase = row0 + mt * 16 + ((lane >> 4) << 3);   // 8-aligned
      int bq = tbase / T_;
      int t  = tbase % T_;
#pragma unroll
      for (int nt = 0; nt < 4; ++nt) {
        int cg = col0 + nt * 16 + (lane & 15);
        int hh = cg >> 7, hd = cg & (HD_ - 1);
        v8bf pk;
#pragma unroll
        for (int v = 0; v < 8; ++v) pk[v] = (bf16)acc[mt][nt][v];
        *(v8bf*)(out + (((size_t)(bq * H_ + hh) * HD_ + hd) * T_ + t)) = pk;
      }
    }
  }
}

// ---------------------------------------------------------------------------
// K2: raw scores S = scale*q.k^T + slope*(j-i), causal -inf, fp32 into attn.
// Grid (T/64, T/256, B*H); block 256 thr, wave tile 32x64. Fully-masked
// blocks/waves skipped (softmax kernel zero-fills those columns).
// ---------------------------------------------------------------------------
__global__ __launch_bounds__(256) void scores_kernel(const bf16* __restrict__ q,
                                                     const bf16* __restrict__ k,
                                                     float* __restrict__ attn) {
  int i0 = blockIdx.x * 64;
  int j0 = blockIdx.y * 256;
  if (j0 > i0 + 63) return;                        // entire block above diagonal
  int bh = blockIdx.z;
  int bq = bh >> 4, h = bh & 15;
  int wid = threadIdx.x >> 5;
  int wm = wid & 1, wn = wid >> 1;
  int row0 = i0 + wm * 32;
  int col0 = j0 + wn * 64;
  if (col0 > row0 + 31) return;                    // wave fully masked
  int lane = threadIdx.x & 31;
  float slope = exp2f(-(float)(h + 1));            // ALiBi slopes, H=16 pow2 path
  const bf16* qbase = q + (size_t)bq * T_ * C_ + h * HD_;
  const bf16* kbase = k + (size_t)bq * T_ * C_ + h * HD_;
  float* arow = attn + (size_t)bh * T_ * T_;
  v8f acc[2][4] = {};
#pragma unroll
  for (int k0 = 0; k0 < HD_; k0 += 32) {
    v16bf a[2];
#pragma unroll
    for (int mt = 0; mt < 2; ++mt)
      a[mt] = load_a_bf16(qbase + (size_t)(row0 + mt * 16) * C_ + k0, C_);
#pragma unroll
    for (int nt = 0; nt < 4; ++nt) {
      v16bf bfrag = load_b_bf16(kbase + (size_t)(col0 + nt * 16) * C_ + k0, C_);
#pragma unroll
      for (int mt = 0; mt < 2; ++mt)
        acc[mt][nt] = wmma_bf16(a[mt], bfrag, acc[mt][nt]);
    }
  }
#pragma unroll
  for (int mt = 0; mt < 2; ++mt) {
    int ib = row0 + mt * 16 + ((lane >> 4) << 3);
#pragma unroll
    for (int nt = 0; nt < 4; ++nt) {
      int j = col0 + nt * 16 + (lane & 15);
#pragma unroll
      for (int v = 0; v < 8; ++v) {
        int i = ib + v;
        float val = (j <= i) ? acc[mt][nt][v] * SCALE + slope * (float)(j - i)
                             : -__builtin_inff();
        arow[(size_t)i * T_ + j] = val;
      }
    }
  }
}

// ---------------------------------------------------------------------------
// K3: in-place row softmax over attn; one wave (32 lanes) per row.
// Writes zeros for the masked tail (also covers poisoned/skipped regions).
// ---------------------------------------------------------------------------
__global__ __launch_bounds__(256) void softmax_kernel(float* __restrict__ attn) {
  int wid = threadIdx.x >> 5;
  int lane = threadIdx.x & 31;
  int row = blockIdx.x * 8 + wid;                  // 0 .. B*H*T-1
  int i = row & (T_ - 1);                          // query index -> valid len i+1
  float* r = attn + (size_t)row * T_;
  float m = -__builtin_inff();
  for (int j = lane; j <= i; j += 32) m = fmaxf(m, r[j]);
#pragma unroll
  for (int s = 16; s > 0; s >>= 1) m = fmaxf(m, __shfl_xor(m, s, 32));
  float sum = 0.f;
  for (int j = lane; j <= i; j += 32) sum += __expf(r[j] - m);
#pragma unroll
  for (int s = 16; s > 0; s >>= 1) sum += __shfl_xor(sum, s, 32);
  float inv = 1.0f / sum;
  for (int j = lane; j < T_; j += 32)
    r[j] = (j <= i) ? __expf(r[j] - m) * inv : 0.0f;
}

// ---------------------------------------------------------------------------
// K4: o_head = attn @ v  (per b,h: M=T, N=HD, K<=row+1 causal).
// A = fp32 attn (cvt to bf16 in-flight, zeros above diagonal keep it exact),
// B = vT (column-major per head -> contiguous 32B B-frag loads).
// Grid (T/128, B*H); block 256 thr, 4 waves M x 2 waves N, wave tile 32x64.
// ---------------------------------------------------------------------------
__global__ __launch_bounds__(256) void av_kernel(const float* __restrict__ attn,
                                                 const bf16* __restrict__ vT,
                                                 bf16* __restrict__ o) {
  int bh = blockIdx.y;
  int bq = bh >> 4, h = bh & 15;
  int wid = threadIdx.x >> 5;
  int wm = wid & 3, wn = wid >> 2;
  int row0 = blockIdx.x * 128 + wm * 32;
  int col0 = wn * 64;
  const float* abase = attn + (size_t)bh * T_ * T_;
  const bf16*  vbase = vT + (size_t)bh * HD_ * T_;
  int kend = row0 + 32;                            // causal trim per wave
  v8f acc[2][4] = {};
#pragma unroll 1
  for (int k0 = 0; k0 < kend; k0 += 32) {
    v16bf a[2];
#pragma unroll
    for (int mt = 0; mt < 2; ++mt)
      a[mt] = load_a_f32(abase + (size_t)(row0 + mt * 16) * T_ + k0, T_);
#pragma unroll
    for (int nt = 0; nt < 4; ++nt) {
      v16bf bfrag = load_b_bf16(vbase + (size_t)(col0 + nt * 16) * T_ + k0, T_);
#pragma unroll
      for (int mt = 0; mt < 2; ++mt)
        acc[mt][nt] = wmma_bf16(a[mt], bfrag, acc[mt][nt]);
    }
  }
#pragma unroll
  for (int mt = 0; mt < 2; ++mt)
#pragma unroll
    for (int nt = 0; nt < 4; ++nt)
      store_d_bf16(o + (size_t)(bq * T_ + row0 + mt * 16) * C_ + h * HD_ +
                       col0 + nt * 16,
                   C_, acc[mt][nt]);
}

// ---------------------------------------------------------------------------
// K5: out = o @ Wo^T + bo, fp32 output. Same tiling as K1.
// ---------------------------------------------------------------------------
__global__ __launch_bounds__(256) void gemm_xwt_f32bias(const bf16* __restrict__ A,
                                                        const bf16* __restrict__ W,
                                                        const float* __restrict__ bias,
                                                        float* __restrict__ out) {
  const int K = C_, N = C_;
  int wid = threadIdx.x >> 5;
  int wm = wid & 1, wn = wid >> 1;
  int row0 = blockIdx.x * 128 + wm * 64;
  int col0 = blockIdx.y * 256 + wn * 64;
  v8f acc[4][4] = {};
#pragma unroll 1
  for (int k0 = 0; k0 < K; k0 += 32) {
    v16bf a[4];
#pragma unroll
    for (int mt = 0; mt < 4; ++mt)
      a[mt] = load_a_bf16(A + (size_t)(row0 + mt * 16) * K + k0, K);
#pragma unroll
    for (int nt = 0; nt < 4; ++nt) {
      v16bf b = load_b_bf16(W + (size_t)(col0 + nt * 16) * K + k0, K);
#pragma unroll
      for (int mt = 0; mt < 4; ++mt)
        acc[mt][nt] = wmma_bf16(a[mt], b, acc[mt][nt]);
    }
  }
  int lane = threadIdx.x & 31;
  int n  = lane & 15;
  int mo = (lane >> 4) << 3;
#pragma unroll
  for (int mt = 0; mt < 4; ++mt)
#pragma unroll
    for (int nt = 0; nt < 4; ++nt) {
      int colg = col0 + nt * 16 + n;
      float bv = bias[colg];
#pragma unroll
      for (int v = 0; v < 8; ++v)
        out[(size_t)(row0 + mt * 16 + mo + v) * N + colg] = acc[mt][nt][v] + bv;
    }
}

// ---------------------------------------------------------------------------
extern "C" void kernel_launch(void* const* d_in, const int* in_sizes, int n_in,
                              void* d_out, int out_size, void* d_ws, size_t ws_size,
                              hipStream_t stream) {
  (void)in_sizes; (void)n_in; (void)out_size; (void)ws_size;
  const float* x  = (const float*)d_in[0];
  const float* Wq = (const float*)d_in[1];
  const float* Wk = (const float*)d_in[2];
  const float* Wv = (const float*)d_in[3];
  const float* Wo = (const float*)d_in[4];
  const float* bo = (const float*)d_in[5];
  float* out  = (float*)d_out;
  float* attn = out + (size_t)B_ * T_ * C_;

  const size_t NX = (size_t)M_ * C_;               // 8,388,608
  const size_t NW = (size_t)C_ * C_;               // 4,194,304
  bf16* p = (bf16*)d_ws;                           // ~112 MB bf16 workspace
  bf16* xb  = p; p += NX;
  bf16* wqb = p; p += NW;
  bf16* wkb = p; p += NW;
  bf16* wvb = p; p += NW;
  bf16* wob = p; p += NW;
  bf16* qb  = p; p += NX;
  bf16* kb  = p; p += NX;
  bf16* vT  = p; p += NX;                          // (B,H,HD,T)
  bf16* ob  = p; p += NX;

  cvt_f32_bf16<<<NX / 1024, 256, 0, stream>>>(x,  xb,  (int)NX);
  cvt_f32_bf16<<<NW / 1024, 256, 0, stream>>>(Wq, wqb, (int)NW);
  cvt_f32_bf16<<<NW / 1024, 256, 0, stream>>>(Wk, wkb, (int)NW);
  cvt_f32_bf16<<<NW / 1024, 256, 0, stream>>>(Wv, wvb, (int)NW);
  cvt_f32_bf16<<<NW / 1024, 256, 0, stream>>>(Wo, wob, (int)NW);

  dim3 gp(M_ / 128, C_ / 256);                     // (32, 8)
  gemm_xwt_bf16<<<gp, 256, 0, stream>>>(xb, wqb, qb, 0);
  gemm_xwt_bf16<<<gp, 256, 0, stream>>>(xb, wkb, kb, 0);
  gemm_xwt_bf16<<<gp, 256, 0, stream>>>(xb, wvb, vT, 1);

  dim3 gs(T_ / 64, T_ / 256, B_ * H_);             // (32, 8, 32)
  scores_kernel<<<gs, 256, 0, stream>>>(qb, kb, attn);

  softmax_kernel<<<(B_ * H_ * T_) / 8, 256, 0, stream>>>(attn);

  dim3 ga(T_ / 128, B_ * H_);                      // (16, 32)
  av_kernel<<<ga, 256, 0, stream>>>(attn, vT, ob);

  gemm_xwt_f32bias<<<gp, 256, 0, stream>>>(ob, wob, bo, out);
}
